// BalRNN_7533372637365
// MI455X (gfx1250) — matrix operand order repaired and checked by reading, not compiled, not run on hardware
//
#include <hip/hip_runtime.h>
#include <hip/hip_bf16.h>
#include <math.h>

// Problem constants (match reference)
#define B_   16
#define S_   2048
#define I_   256
#define H_   1024
#define NWG  16           // persistent RNN workgroups (one per WGP)
#define KT_IH (I_/32)     // 8 k-tiles for input GEMM
#define KT_HH (H_/32)     // 32 k-tiles for hidden GEMM

typedef __attribute__((ext_vector_type(16))) __bf16   v16bf;
typedef __attribute__((ext_vector_type(8)))  float    v8f;
typedef __attribute__((ext_vector_type(4)))  unsigned u32x4;

union V16 { u32x4 u[2]; v16bf v; };

__device__ __forceinline__ unsigned short f2bf(float f) {
  union { float f; unsigned u; } x; x.f = f;
  unsigned r = x.u + 0x7fffu + ((x.u >> 16) & 1u);   // round-to-nearest-even
  return (unsigned short)(r >> 16);
}
__device__ __forceinline__ float bf2f(unsigned short b) {
  union { unsigned u; float f; } x; x.u = ((unsigned)b) << 16;
  return x.f;
}

// ---- A-operand load: 16 bf16 per lane from a contiguous bf16 row ----
// 16-bit A 16x32 layout: lanes 0-15 hold K {0..7, 16..23}; lanes 16-31 hold K {8..15, 24..31}
__device__ __forceinline__ v16bf load_a_row(const unsigned short* rowbase, int k0, int lane) {
  const char* p = (const char*)rowbase + k0 * 2 + ((lane & 16) ? 16 : 0);
  V16 t;
  t.u[0] = *(const u32x4*)(p);
  t.u[1] = *(const u32x4*)(p + 32);
  return t.v;
}

__device__ __forceinline__ v16bf load_b_lds(const char* base) {
  const u32x4* bp = (const u32x4*)base;
  V16 b; b.u[0] = bp[0]; b.u[1] = bp[1];
  return b.v;
}

// ---- grid-wide sense barrier over NWG workgroups ----
__device__ __forceinline__ void grid_barrier(unsigned* counter, unsigned target) {
  __syncthreads();
  if (threadIdx.x == 0) {
    __hip_atomic_fetch_add(counter, 1u, __ATOMIC_RELEASE, __HIP_MEMORY_SCOPE_AGENT);
    while (__hip_atomic_load(counter, __ATOMIC_ACQUIRE, __HIP_MEMORY_SCOPE_AGENT) < target) {
      __builtin_amdgcn_s_sleep(1);
    }
  }
  __syncthreads();
}

// =====================================================================
// Prep: fp32 -> bf16 stream conversion (for x)
// =====================================================================
__global__ void cvt_bf16_kernel(const float* __restrict__ src,
                                unsigned short* __restrict__ dst, int n) {
  int i = blockIdx.x * blockDim.x + threadIdx.x;
  if (i < n) dst[i] = f2bf(src[i]);
}

// =====================================================================
// Prep: pack W[N][K] (row major fp32) into per-lane WMMA B tiles (bf16)
// tile = (ntile, ktile) of 16(N) x 32(K); 1 KB per tile, 32 B per lane.
// B layout: vgpr j: K=8h+2j,+1 (j<4), K=8+8h+2j,+1 (j>=4), h=lane/16 @ N=lane%16
// =====================================================================
__global__ void pack_b_kernel(const float* __restrict__ W,
                              unsigned* __restrict__ dst, int N, int K) {
  int nk = K / 32;
  int total = (N / 16) * nk * 256;           // dwords
  int tid = blockIdx.x * blockDim.x + threadIdx.x;
  if (tid >= total) return;
  int tile = tid >> 8, r = tid & 255;
  int lane = r >> 3, j = r & 7;
  int ntile = tile / nk, ktile = tile % nk;
  int n  = ntile * 16 + (lane & 15);
  int kk = ktile * 32 + 2 * j + ((lane >= 16) ? 8 : 0) + ((j >= 4) ? 8 : 0);
  unsigned short lo = f2bf(W[(size_t)n * K + kk]);
  unsigned short hi = f2bf(W[(size_t)n * K + kk + 1]);
  dst[tid] = (unsigned)lo | ((unsigned)hi << 16);
}

// =====================================================================
// Persistent RNN kernel: 16 WGs x 128 threads (4 waves). Wave w of WG g
// owns hidden tile ntile = g*4+w (16 neurons). All weight slices live in LDS.
// Dual accumulator chains per K-loop hide XDL WMMA latency (1 wave/SIMD).
// =====================================================================
__global__ __launch_bounds__(128)
void rnn_kernel(const unsigned short* __restrict__ xbf,
                const unsigned* __restrict__ wih_p,
                const unsigned* __restrict__ whh0_p,
                const unsigned* __restrict__ whh1_p,
                unsigned short* h0buf,     // double-buffered [2][B][H]
                unsigned short* h1buf,     // [B][H]
                unsigned short* sbuf,      // [B][H]
                unsigned short* outs,      // [B][S][H]
                unsigned* counter) {
  extern __shared__ char smem[];
  const int tid  = threadIdx.x;
  const int lane = tid & 31;
  const int wave = tid >> 5;
  const int wg   = blockIdx.x;

  // LDS: wih slice 32KB | whh0 slice 128KB | whh1 slice 128KB  (288KB total)
  char* wih_l  = smem;
  char* whh0_l = smem + 32 * 1024;
  char* whh1_l = smem + (32 + 128) * 1024;

  // Preload this WG's weight slices (contiguous in packed layout)
  {
    const u32x4* s0 = (const u32x4*)wih_p  + (size_t)wg * (32  * 1024 / 16);
    const u32x4* s1 = (const u32x4*)whh0_p + (size_t)wg * (128 * 1024 / 16);
    const u32x4* s2 = (const u32x4*)whh1_p + (size_t)wg * (128 * 1024 / 16);
    u32x4* d0 = (u32x4*)wih_l;  u32x4* d1 = (u32x4*)whh0_l;  u32x4* d2 = (u32x4*)whh1_l;
    for (int i = tid; i < 32  * 1024 / 16; i += 128) d0[i] = s0[i];
    for (int i = tid; i < 128 * 1024 / 16; i += 128) d1[i] = s1[i];
    for (int i = tid; i < 128 * 1024 / 16; i += 128) d2[i] = s2[i];
  }
  __syncthreads();

  const int ln    = wave;                    // local n-tile 0..3
  const int ntile = wg * 4 + ln;
  const int n0    = ntile * 16;
  const float cadd = 3.1622776601683795f;    // sqrt(K=10)

  const int row = lane & 15;                 // batch row this lane serves for A
  const unsigned short* xrow = xbf + (size_t)row * S_ * I_;

  unsigned bar = 0;
  for (int t = 0; t < S_; ++t) {
    const unsigned short* h0cur = h0buf + (size_t)(t & 1) * (B_ * H_);
    unsigned short*       h0nxt = h0buf + (size_t)((t + 1) & 1) * (B_ * H_);

    // ---- layer 0: ff(x_t) + recurrent(h0_prev), two independent chains ----
    v8f accA = {}, accB = {};
    const unsigned short* xtrow = xrow + (size_t)t * I_;
#pragma unroll
    for (int kt = 0; kt < KT_IH; kt += 2) {
      v16bf a0 = load_a_row(xtrow, kt * 32, lane);
      v16bf a1 = load_a_row(xtrow, (kt + 1) * 32, lane);
      v16bf b0 = load_b_lds(wih_l + ((ln * KT_IH + kt)     * 1024 + lane * 32));
      v16bf b1 = load_b_lds(wih_l + ((ln * KT_IH + kt + 1) * 1024 + lane * 32));
      accA = __builtin_amdgcn_wmma_f32_16x16x32_bf16(false, a0, false, b0, (short)0, accA, false, false);
      accB = __builtin_amdgcn_wmma_f32_16x16x32_bf16(false, a1, false, b1, (short)0, accB, false, false);
    }
    const unsigned short* h0row = h0cur + row * H_;
    for (int kt = 0; kt < KT_HH; kt += 2) {
      v16bf a0 = load_a_row(h0row, kt * 32, lane);
      v16bf a1 = load_a_row(h0row, (kt + 1) * 32, lane);
      v16bf b0 = load_b_lds(whh0_l + ((ln * KT_HH + kt)     * 1024 + lane * 32));
      v16bf b1 = load_b_lds(whh0_l + ((ln * KT_HH + kt + 1) * 1024 + lane * 32));
      accA = __builtin_amdgcn_wmma_f32_16x16x32_bf16(false, a0, false, b0, (short)0, accA, false, false);
      accB = __builtin_amdgcn_wmma_f32_16x16x32_bf16(false, a1, false, b1, (short)0, accB, false, false);
    }
    v8f acc = accA + accB;
    // activation; publish h0_new and s = h0_new + h1_prev for this slice
    {
      int nn = n0 + (lane & 15);
      int mbase = (lane & 16) ? 8 : 0;
#pragma unroll
      for (int r = 0; r < 8; ++r) {
        int m = mbase + r;
        float h0v = fmaxf(cadd + acc[r], 0.0f);
        h0nxt[m * H_ + nn] = f2bf(h0v);
        float sv = h0v + bf2f(h1buf[m * H_ + nn]);
        sbuf[m * H_ + nn] = f2bf(sv);
      }
    }
    if (t + 1 < S_) __builtin_prefetch(xrow + (size_t)(t + 1) * I_, 0, 0);

    grid_barrier(counter, ++bar * NWG);      // s fully published

    // ---- layer 1: (h0' + h1) @ Whh1^T, two independent chains ----
    v8f acc1A = {}, acc1B = {};
    const unsigned short* srow = sbuf + row * H_;
    for (int kt = 0; kt < KT_HH; kt += 2) {
      v16bf a0 = load_a_row(srow, kt * 32, lane);
      v16bf a1 = load_a_row(srow, (kt + 1) * 32, lane);
      v16bf b0 = load_b_lds(whh1_l + ((ln * KT_HH + kt)     * 1024 + lane * 32));
      v16bf b1 = load_b_lds(whh1_l + ((ln * KT_HH + kt + 1) * 1024 + lane * 32));
      acc1A = __builtin_amdgcn_wmma_f32_16x16x32_bf16(false, a0, false, b0, (short)0, acc1A, false, false);
      acc1B = __builtin_amdgcn_wmma_f32_16x16x32_bf16(false, a1, false, b1, (short)0, acc1B, false, false);
    }
    v8f acc1 = acc1A + acc1B;
    {
      int nn = n0 + (lane & 15);
      int mbase = (lane & 16) ? 8 : 0;
#pragma unroll
      for (int r = 0; r < 8; ++r) {
        int m = mbase + r;
        float h1v = fmaxf(cadd + acc1[r], 0.0f);
        unsigned short h1b = f2bf(h1v);
        h1buf[m * H_ + nn] = h1b;
        outs[((size_t)m * S_ + t) * H_ + nn] = h1b;
      }
    }
    grid_barrier(counter, ++bar * NWG);      // step complete
  }
}

// =====================================================================
// Output projection: out[B*S,256] = outs[B*S,1024](bf16) @ out_w^T + b
// =====================================================================
__global__ __launch_bounds__(128)
void proj_kernel(const unsigned short* __restrict__ outs,
                 const unsigned* __restrict__ outw_p,
                 const float* __restrict__ out_b,
                 float* __restrict__ out) {
  const int tid = threadIdx.x, lane = tid & 31, wave = tid >> 5;
  const int mtile = blockIdx.x >> 2;                 // 0..2047
  const int ntile = (blockIdx.x & 3) * 4 + wave;     // 0..15
  const int n0 = ntile * 16;
  const int m0 = mtile * 16;
  const unsigned short* arow = outs + (size_t)(m0 + (lane & 15)) * H_;
  v8f accA = {}, accB = {};
  for (int kt = 0; kt < H_ / 32; kt += 2) {
    v16bf a0 = load_a_row(arow, kt * 32, lane);
    v16bf a1 = load_a_row(arow, (kt + 1) * 32, lane);
    const u32x4* bp0 = (const u32x4*)outw_p + (((size_t)(ntile * (H_ / 32) + kt)) * 32 + lane) * 2;
    const u32x4* bp1 = (const u32x4*)outw_p + (((size_t)(ntile * (H_ / 32) + kt + 1)) * 32 + lane) * 2;
    V16 b0; b0.u[0] = bp0[0]; b0.u[1] = bp0[1];
    V16 b1; b1.u[0] = bp1[0]; b1.u[1] = bp1[1];
    accA = __builtin_amdgcn_wmma_f32_16x16x32_bf16(false, a0, false, b0.v, (short)0, accA, false, false);
    accB = __builtin_amdgcn_wmma_f32_16x16x32_bf16(false, a1, false, b1.v, (short)0, accB, false, false);
  }
  v8f acc = accA + accB;
  int nn = n0 + (lane & 15);
  float bias = out_b[nn];
  int mbase = (lane & 16) ? 8 : 0;
#pragma unroll
  for (int r = 0; r < 8; ++r) {
    int m = m0 + mbase + r;
    out[(size_t)m * I_ + nn] = acc[r] + bias;
  }
}

// =====================================================================
// Final hidden state h_T [L=2][B][H] in fp32 appended after out
// =====================================================================
__global__ void hT_kernel(const unsigned short* __restrict__ h0f,
                          const unsigned short* __restrict__ h1f,
                          float* __restrict__ dst) {
  int i = blockIdx.x * blockDim.x + threadIdx.x;
  if (i >= 2 * B_ * H_) return;
  dst[i] = (i < B_ * H_) ? bf2f(h0f[i]) : bf2f(h1f[i - B_ * H_]);
}

// =====================================================================
extern "C" void kernel_launch(void* const* d_in, const int* in_sizes, int n_in,
                              void* d_out, int out_size, void* d_ws, size_t ws_size,
                              hipStream_t stream) {
  (void)in_sizes; (void)n_in; (void)out_size; (void)ws_size;
  const float* x      = (const float*)d_in[0];   // [16,2048,256]
  const float* w_ih   = (const float*)d_in[1];   // [2,1024,256] (layer0 used)
  const float* w_hh   = (const float*)d_in[2];   // [2,1024,1024]
  const float* out_w  = (const float*)d_in[3];   // [256,1024]
  const float* out_b  = (const float*)d_in[4];   // [256]
  float* out = (float*)d_out;

  // workspace layout (bytes)
  char* ws = (char*)d_ws;
  size_t off = 0;
  unsigned short* xbf    = (unsigned short*)(ws + off); off += (size_t)B_ * S_ * I_ * 2;     // 16 MB
  unsigned*       wih_p  = (unsigned*)(ws + off);       off += (size_t)(H_/16)*(I_/32)*1024; // 512 KB
  unsigned*       whh0_p = (unsigned*)(ws + off);       off += (size_t)(H_/16)*(H_/32)*1024; // 2 MB
  unsigned*       whh1_p = (unsigned*)(ws + off);       off += (size_t)(H_/16)*(H_/32)*1024; // 2 MB
  unsigned*       outw_p = (unsigned*)(ws + off);       off += (size_t)(I_/16)*(H_/32)*1024; // 512 KB
  unsigned short* outs   = (unsigned short*)(ws + off); off += (size_t)B_ * S_ * H_ * 2;     // 64 MB
  unsigned short* h0buf  = (unsigned short*)(ws + off); off += 2 * (size_t)B_ * H_ * 2;
  unsigned short* h1buf  = (unsigned short*)(ws + off); off += (size_t)B_ * H_ * 2;
  unsigned short* sbuf   = (unsigned short*)(ws + off); off += (size_t)B_ * H_ * 2;
  unsigned*       cnt    = (unsigned*)(ws + off);       off += 256;

  // zero state + barrier (graph-capture-safe)
  hipMemsetAsync(h0buf, 0, 2 * (size_t)B_ * H_ * 2, stream);
  hipMemsetAsync(h1buf, 0, (size_t)B_ * H_ * 2, stream);
  hipMemsetAsync(cnt,   0, 256, stream);

  // prep
  {
    int n = B_ * S_ * I_;
    cvt_bf16_kernel<<<(n + 255) / 256, 256, 0, stream>>>(x, xbf, n);
  }
  pack_b_kernel<<<((H_/16)*(I_/32)*256 + 255) / 256, 256, 0, stream>>>(w_ih, wih_p, H_, I_);
  pack_b_kernel<<<((H_/16)*(H_/32)*256 + 255) / 256, 256, 0, stream>>>(w_hh, whh0_p, H_, H_);
  pack_b_kernel<<<((H_/16)*(H_/32)*256 + 255) / 256, 256, 0, stream>>>(w_hh + (size_t)H_ * H_, whh1_p, H_, H_);
  pack_b_kernel<<<((I_/16)*(H_/32)*256 + 255) / 256, 256, 0, stream>>>(out_w, outw_p, I_, H_);

  // persistent RNN with 288 KB dynamic LDS per WG (CDNA5: 320 KB available)
  const int lds_bytes = (32 + 128 + 128) * 1024;
  (void)hipFuncSetAttribute((const void*)rnn_kernel,
                            hipFuncAttributeMaxDynamicSharedMemorySize, lds_bytes);
  rnn_kernel<<<NWG, 128, lds_bytes, stream>>>(xbf, wih_p, whh0_p, whh1_p,
                                              h0buf, h1buf, sbuf, outs, cnt);

  // output projection and h_T tail
  proj_kernel<<<(B_ * S_ / 16) * 4, 128, 0, stream>>>(outs, outw_p, out_b, out);
  hT_kernel<<<(2 * B_ * H_ + 255) / 256, 256, 0, stream>>>(h0buf /*parity 0 = final*/, h1buf,
                                                           out + (size_t)B_ * S_ * I_);
}
